// SWMSA_81819126989590
// MI455X (gfx1250) — matrix operand order repaired and checked by reading, not compiled, not run on hardware
//
#include <hip/hip_runtime.h>
#include <hip/hip_bf16.h>

typedef __attribute__((ext_vector_type(16))) _Float16 v16h;
typedef __attribute__((ext_vector_type(8)))  float    v8f;

#define PITCH 66   // LDS row pitch in halves (odd bank stride -> conflict-free)

// CDNA5 async global->LDS path (ASYNCcnt), guarded so either toolchain compiles.
#if defined(__gfx1250__) && \
    __has_builtin(__builtin_amdgcn_global_load_async_to_lds_b128) && \
    __has_builtin(__builtin_amdgcn_s_wait_asynccnt)
#define HAS_ASYNC_LDS 1
#else
#define HAS_ASYNC_LDS 0
#endif

#if HAS_ASYNC_LDS
// builtin prototype uses 16-byte int vectors: (v4i as1* src, v4i as3* dst, imm, imm)
typedef int b128_t __attribute__((vector_size(16)));
typedef __attribute__((address_space(1))) b128_t* gptr_b128;
typedef __attribute__((address_space(3))) b128_t* lptr_b128;
#endif

__device__ __forceinline__ v8f wmma_f16(v16h a, v16h b, v8f c) {
  // D = A*B + C, f16 inputs, f32 accumulate  -> v_wmma_f32_16x16x32_f16
  return __builtin_amdgcn_wmma_f32_16x16x32_f16(false, a, false, b, (short)0, c,
                                                false, false);
}

// A fragment (16x32 f16): m = lane%16; K = hi*16 + idx
__device__ __forceinline__ v16h fragA(const _Float16* p, int pitch, int row,
                                      int kbase, int hi) {
  v16h a;
  const _Float16* rp = p + row * pitch + kbase + hi * 16;
#pragma unroll
  for (int i = 0; i < 8; ++i) a[i] = rp[i];
#pragma unroll
  for (int i = 8; i < 16; ++i) a[i] = rp[i];
  return a;
}

// B fragment (32x16 f16) from row-major [n][k] storage: n = lane%16, K = hi*16 + idx
__device__ __forceinline__ v16h fragB_rowmajor(const _Float16* p, int pitch,
                                               int nrow, int kbase, int hi) {
  v16h b;
  const _Float16* rp = p + nrow * pitch + kbase + hi * 16;
#pragma unroll
  for (int i = 0; i < 16; ++i) b[i] = rp[i];
  return b;
}

// 16-byte global->LDS copy: async DMA on gfx1250 toolchains that expose it.
__device__ __forceinline__ void copy8h_g2l(const _Float16* g, _Float16* l) {
#if HAS_ASYNC_LDS
  __builtin_amdgcn_global_load_async_to_lds_b128((gptr_b128)g, (lptr_b128)l, 0, 0);
#else
#pragma unroll
  for (int i = 0; i < 8; ++i) l[i] = g[i];
#endif
}

// ---------------------------------------------------------------------------
// Kernel 1: roll(-4,-4) + window partition + per-head QKV + attention.
// One wave (32 threads) per (batch, window, head). Output merged-head f16.
// ---------------------------------------------------------------------------
__global__ void __launch_bounds__(32)
swin_attn_kernel(const float* __restrict__ x, const float* __restrict__ Wqkv,
                 const float* __restrict__ bqkv, _Float16* __restrict__ tmid) {
  __shared__ _Float16 Xs[64 * PITCH];  // X_h tile [64 tok][64 d]; reused as P strip
  __shared__ _Float16 Qs[64 * PITCH];
  __shared__ _Float16 Ks[64 * PITCH];
  __shared__ _Float16 Vs[64 * PITCH];

  const int lane = threadIdx.x & 31;
  const int n16  = lane & 15;
  const int hi   = (lane >> 4) & 1;
  const int w    = blockIdx.x >> 3;   // window 0..255
  const int head = blockIdx.x & 7;
  const int b    = blockIdx.y;
  const int wh = w >> 4, ww = w & 15;

  // ---- stage X_h with fused cyclic shift: xr[r][c] = x[(r+4)%128][(c+4)%128]
  for (int t = lane; t < 64; t += 32) {
    const int r = ((wh << 3) + (t >> 3) + 4) & 127;
    const int c = ((ww << 3) + (t & 7) + 4) & 127;
    const float* src = x + ((size_t)b * 16384 + r * 128 + c) * 512 + head * 64;
#pragma unroll
    for (int d = 0; d < 64; d += 4) {
      float4 v4 = *(const float4*)(src + d);
      Xs[t * PITCH + d + 0] = (_Float16)v4.x;
      Xs[t * PITCH + d + 1] = (_Float16)v4.y;
      Xs[t * PITCH + d + 2] = (_Float16)v4.z;
      Xs[t * PITCH + d + 3] = (_Float16)v4.w;
    }
  }
  __syncthreads();

  // ---- QKV: [64 x 64] * W_qkv^T [64 x 192] + bias  (bias preloaded into C)
  for (int nt = 0; nt < 12; ++nt) {
    v16h b0, b1;
#pragma unroll
    for (int i = 0; i < 16; ++i) {
      // B[k=d][n=e_out] = W_qkv[e_out][d]; row-major read from global (L2-hot)
      b0[i] = (_Float16)Wqkv[(nt * 16 + n16) * 64 + 0  + hi * 16 + i];
      b1[i] = (_Float16)Wqkv[(nt * 16 + n16) * 64 + 32 + hi * 16 + i];
    }
    const float bias = bqkv[nt * 16 + n16];
    _Float16* dst = (nt < 4) ? Qs : (nt < 8) ? Ks : Vs;
    const int ncol = (nt & 3) * 16 + n16;
    for (int mt = 0; mt < 4; ++mt) {
      v8f c;
#pragma unroll
      for (int i = 0; i < 8; ++i) c[i] = bias;
      c = wmma_f16(fragA(Xs, PITCH, mt * 16 + n16, 0,  hi), b0, c);
      c = wmma_f16(fragA(Xs, PITCH, mt * 16 + n16, 32, hi), b1, c);
#pragma unroll
      for (int rr = 0; rr < 8; ++rr)  // D layout: m = hi*8+rr, n = n16
        dst[(mt * 16 + hi * 8 + rr) * PITCH + ncol] = (_Float16)c[rr];
    }
  }
  __syncthreads();

  // ---- attention, one 16-row strip at a time (keeps S in registers)
  _Float16* Ps = Xs;  // X tile is dead now; reuse for P (16x64 strip)
  const float scale = 0.125f;  // 1/sqrt(64)
  for (int mt = 0; mt < 4; ++mt) {
    v8f s[4];
    for (int nt = 0; nt < 4; ++nt) {
      v8f c;
#pragma unroll
      for (int i = 0; i < 8; ++i) c[i] = 0.0f;
      // S = Q * K^T : B[k=d][n=key] = K[key][d] -> row-major read of Ks
      c = wmma_f16(fragA(Qs, PITCH, mt * 16 + n16, 0,  hi),
                   fragB_rowmajor(Ks, PITCH, nt * 16 + n16, 0,  hi), c);
      c = wmma_f16(fragA(Qs, PITCH, mt * 16 + n16, 32, hi),
                   fragB_rowmajor(Ks, PITCH, nt * 16 + n16, 32, hi), c);
#pragma unroll
      for (int i = 0; i < 8; ++i) c[i] *= scale;
      s[nt] = c;
    }
    // softmax over 64 keys; row m = mt*16 + hi*8 + rr lives on 16 lanes (same hi)
#pragma unroll
    for (int rr = 0; rr < 8; ++rr) {
      float mx = s[0][rr];
#pragma unroll
      for (int nt = 1; nt < 4; ++nt) mx = fmaxf(mx, s[nt][rr]);
#pragma unroll
      for (int off = 1; off < 16; off <<= 1)
        mx = fmaxf(mx, __shfl_xor(mx, off, 32));  // xor of bits 0..3 stays in half
      float e0 = __expf(s[0][rr] - mx);
      float e1 = __expf(s[1][rr] - mx);
      float e2 = __expf(s[2][rr] - mx);
      float e3 = __expf(s[3][rr] - mx);
      float sum = e0 + e1 + e2 + e3;
#pragma unroll
      for (int off = 1; off < 16; off <<= 1) sum += __shfl_xor(sum, off, 32);
      const float inv = __builtin_amdgcn_rcpf(sum);
      // C-layout -> A-layout transpose through LDS
      Ps[(hi * 8 + rr) * PITCH +  0 + n16] = (_Float16)(e0 * inv);
      Ps[(hi * 8 + rr) * PITCH + 16 + n16] = (_Float16)(e1 * inv);
      Ps[(hi * 8 + rr) * PITCH + 32 + n16] = (_Float16)(e2 * inv);
      Ps[(hi * 8 + rr) * PITCH + 48 + n16] = (_Float16)(e3 * inv);
    }
    // O strip = P (16x64) * V (64x64); B[k=key][n=d] = V[key][d]
    for (int dt = 0; dt < 4; ++dt) {
      v8f c;
#pragma unroll
      for (int i = 0; i < 8; ++i) c[i] = 0.0f;
#pragma unroll
      for (int kt = 0; kt < 2; ++kt) {
        v16h a = fragA(Ps, PITCH, n16, kt * 32, hi);
        v16h bb;
#pragma unroll
        for (int i = 0; i < 16; ++i)
          bb[i] = Vs[(kt * 32 + hi * 16 + i) * PITCH + dt * 16 + n16];
        c = wmma_f16(a, bb, c);
      }
#pragma unroll
      for (int rr = 0; rr < 8; ++rr) {
        const int tok = w * 64 + mt * 16 + hi * 8 + rr;  // raw-reshape token order
        tmid[((size_t)b * 16384 + tok) * 512 + head * 64 + dt * 16 + n16] =
            (_Float16)c[rr];
      }
    }
  }
}

// ---------------------------------------------------------------------------
// Kernel 2: roll(+4,+4) fused into gather + final projection GEMM.
// Block tile: 128 tokens (one output image row) x 128 out-channels; 8 waves,
// each wave 32x64 (2x4 WMMA tiles). K streamed in 32-wide chunks, DOUBLE
// BUFFERED: the A-tile (f16 intermediate) goes global->LDS via async DMA
// (ASYNCcnt) so the next chunk's copy overlaps the current chunk's WMMAs;
// the B-tile (W_proj) needs f32->f16 conversion so it stages through VALU.
// ---------------------------------------------------------------------------
__global__ void __launch_bounds__(256)
proj_kernel(const _Float16* __restrict__ tmid, const float* __restrict__ Wp,
            const float* __restrict__ bp, float* __restrict__ y) {
  __shared__ _Float16 As[2][128][40];  // tokens x k-chunk (double buffered)
  __shared__ _Float16 Bs[2][128][40];  // out-channel x k-chunk (B[k][n]=Wp[n][k])

  const int tid  = threadIdx.x;
  const int lane = tid & 31;
  const int n16  = lane & 15;
  const int hi   = (lane >> 4) & 1;
  const int wave = tid >> 5;
  const int wm = wave >> 1;                 // 0..3 : 32-row band
  const int wn = wave & 1;                  // 0..1 : 64-col band
  const int b  = blockIdx.x >> 7;           // batch
  const int r  = blockIdx.x & 127;          // output image row
  const int fbase = blockIdx.y << 7;        // output channel base
  const int rs = (r - 4) & 127;             // reverse cyclic shift (row)

  const int srow  = tid >> 1;               // staging row 0..127
  const int spart = (tid & 1) << 4;         // 0 or 16 halves (32B)
  const int cs = (srow - 4) & 127;          // reverse cyclic shift (col)
  const _Float16* tsrc = tmid + ((size_t)b * 16384 + rs * 128 + cs) * 512 + spart;
  const float*    wsrc = Wp + (size_t)(fbase + srow) * 512 + spart;

  v8f acc[2][4];
#pragma unroll
  for (int mt = 0; mt < 2; ++mt)
#pragma unroll
    for (int nt = 0; nt < 4; ++nt)
#pragma unroll
      for (int i = 0; i < 8; ++i) acc[mt][nt][i] = 0.0f;

  // prologue: chunk 0 in flight
  copy8h_g2l(tsrc + 0, &As[0][srow][spart + 0]);
  copy8h_g2l(tsrc + 8, &As[0][srow][spart + 8]);
#pragma unroll
  for (int i = 0; i < 16; ++i) Bs[0][srow][spart + i] = (_Float16)wsrc[i];

  for (int kc = 0; kc < 16; ++kc) {
    const int cur = kc & 1;
    const int nxt = cur ^ 1;
    if (kc < 15) {
      const int kb = (kc + 1) * 32;
      copy8h_g2l(tsrc + kb + 0, &As[nxt][srow][spart + 0]);
      copy8h_g2l(tsrc + kb + 8, &As[nxt][srow][spart + 8]);
      if (kc < 14) __builtin_prefetch(wsrc + kb + 32, 0, 0);  // global_prefetch_b8
#pragma unroll
      for (int i = 0; i < 16; ++i)
        Bs[nxt][srow][spart + i] = (_Float16)wsrc[kb + i];
#if HAS_ASYNC_LDS
      // allow the 2 just-issued (next-chunk) async loads to stay in flight;
      // in-order completion => all older (current-chunk) loads have landed
      __builtin_amdgcn_s_wait_asynccnt(2);
#endif
    } else {
#if HAS_ASYNC_LDS
      __builtin_amdgcn_s_wait_asynccnt(0);
#endif
    }
    __syncthreads();  // everyone's async + VALU staging of `cur` is visible
#pragma unroll
    for (int mt = 0; mt < 2; ++mt) {
      v16h a = fragA(&As[cur][0][0], 40, wm * 32 + mt * 16 + n16, 0, hi);
#pragma unroll
      for (int nt = 0; nt < 4; ++nt) {
        v16h bb = fragB_rowmajor(&Bs[cur][0][0], 40, wn * 64 + nt * 16 + n16, 0, hi);
        acc[mt][nt] = wmma_f16(a, bb, acc[mt][nt]);
      }
    }
    __syncthreads();  // done reading `cur` before it is restaged at kc+2
  }

#pragma unroll
  for (int mt = 0; mt < 2; ++mt)
#pragma unroll
    for (int nt = 0; nt < 4; ++nt) {
      const int f = fbase + wn * 64 + nt * 16 + n16;
      const float bias = bp[f];
#pragma unroll
      for (int rr = 0; rr < 8; ++rr) {
        const int c = wm * 32 + mt * 16 + hi * 8 + rr;
        y[((size_t)b * 16384 + r * 128 + c) * 512 + f] = acc[mt][nt][rr] + bias;
      }
    }
}

// ---------------------------------------------------------------------------
extern "C" void kernel_launch(void* const* d_in, const int* in_sizes, int n_in,
                              void* d_out, int out_size, void* d_ws, size_t ws_size,
                              hipStream_t stream) {
  const float* x  = (const float*)d_in[0];   // [16,16384,512]
  const float* Wq = (const float*)d_in[1];   // [192,64]
  const float* bq = (const float*)d_in[2];   // [192]
  const float* Wp = (const float*)d_in[3];   // [512,512]
  const float* bp = (const float*)d_in[4];   // [512]
  float* y = (float*)d_out;                  // [16,16384,512] f32
  _Float16* tmid = (_Float16*)d_ws;          // 16*16384*512 f16 = 256 MB scratch

  // 256 windows * 8 heads blocks per batch, one wave32 per (window, head)
  swin_attn_kernel<<<dim3(2048, 16), 32, 0, stream>>>(x, Wq, bq, tmid);
  // 16 batches * 128 rows, 4 channel tiles of 128
  proj_kernel<<<dim3(2048, 4), 256, 0, stream>>>(tmid, Wp, bp, y);
}
